// DAWNLayer_15668040696248
// MI455X (gfx1250) — compile-verified, matrix operand order checked by reading, not run on hardware
//
#include <hip/hip_runtime.h>
#include <hip/hip_bf16.h>

// ---------------- constants ----------------
#define BB 2
#define SS 512
#define HH 768
#define BS 1024          // B*S
#define NIN 64
#define NPROC 128
#define AR 16
#define PR 128
#define NEG_INF (-1e30f)

typedef __attribute__((ext_vector_type(16))) __bf16 v16bf;
typedef __attribute__((ext_vector_type(8)))  __bf16 v8bf;
typedef __attribute__((ext_vector_type(8)))  float  v8f;

// ---------------- helpers ----------------
__device__ inline __bf16 f2bf(float f) {
    unsigned int u = __builtin_bit_cast(unsigned int, f);
    unsigned short r = (unsigned short)((u + 0x7FFFu + ((u >> 16) & 1u)) >> 16);
    return __builtin_bit_cast(__bf16, r);
}
__device__ inline v8f wmma_bf16(v16bf a, v16bf b, v8f c) {
    // 8 args: (neg_a, A, neg_b, B, c_mod, C, reuse_a, reuse_b)
    return __builtin_amdgcn_wmma_f32_16x16x32_bf16(false, a, false, b, (short)0, c, false, false);
}
// Per ISA 7.12.2 a lane's 16 bf16 fragment elements are two contiguous runs of 8:
// k = 8*hi .. 8*hi+7 and k = 16+8*hi .. 16+8*hi+7  ->  two 16-byte loads.
__device__ inline v16bf ldfrag(const __bf16* p) {
    v8bf lo  = *(const v8bf*)p;
    v8bf hi8 = *(const v8bf*)(p + 16);
    return __builtin_shufflevector(lo, hi8, 0,1,2,3,4,5,6,7,8,9,10,11,12,13,14,15);
}
__device__ inline float sigmoidf(float x) { return 1.0f / (1.0f + expf(-x)); }

// ---------------- vectorized bf16 conversion / layout kernels ----------------
// plain: dst[i] = bf16(src[i]), 8 per thread
__global__ void k_cvt8(const float* __restrict__ src, __bf16* __restrict__ dst, int n8) {
    int i = blockIdx.x * 256 + threadIdx.x;
    if (i < n8) {
        const float4* s = (const float4*)src + (size_t)i * 2;
        float4 a = s[0], b = s[1];
        v8bf d;
        d[0]=f2bf(a.x); d[1]=f2bf(a.y); d[2]=f2bf(a.z); d[3]=f2bf(a.w);
        d[4]=f2bf(b.x); d[5]=f2bf(b.y); d[6]=f2bf(b.z); d[7]=f2bf(b.w);
        *(v8bf*)(dst + (size_t)i * 8) = d;
    }
}
// generic 2D transpose-cvt: dst[r*C + c] = src[c*R + r], vector along c
__global__ void k_cvtT8(const float* __restrict__ src, __bf16* __restrict__ dst, int R, int C) {
    int i = blockIdx.x * 256 + threadIdx.x;
    if (i < R * C / 8) {
        int idx = i * 8, r = idx / C, c0 = idx % C;
        v8bf d;
#pragma unroll
        for (int j = 0; j < 8; ++j) d[j] = f2bf(src[(size_t)(c0 + j) * R + r]);
        *(v8bf*)(dst + idx) = d;
    }
}
// adapt_down (n,k,r) -> adBT[(n*16+r)*768 + k], vector along k
__global__ void k_cvt_adT8(const float* __restrict__ ad, __bf16* __restrict__ dst) {
    int i = blockIdx.x * 256 + threadIdx.x;
    if (i < 1024 * HH / 8) {
        int idx = i * 8, col = idx / HH, k0 = idx % HH;
        int n = col >> 4, r = col & 15;
        const float* s = ad + (size_t)n * HH * AR + (size_t)k0 * AR + r;
        v8bf d;
#pragma unroll
        for (int j = 0; j < 8; ++j) d[j] = f2bf(s[(size_t)j * AR]);
        *(v8bf*)(dst + idx) = d;
    }
}
// down_proj (n,k,r) -> dpbT[n][r*768 + k], vector along k
__global__ void k_cvt_dpT8(const float* __restrict__ dp, __bf16* __restrict__ dst) {
    int i = blockIdx.x * 256 + threadIdx.x;
    if (i < NPROC * PR * HH / 8) {
        int idx = i * 8;
        int n = idx / (PR * HH), rem = idx % (PR * HH);
        int r = rem / HH, k0 = rem % HH;
        const float* s = dp + (size_t)n * HH * PR + (size_t)k0 * PR + r;
        v8bf d;
#pragma unroll
        for (int j = 0; j < 8; ++j) d[j] = f2bf(s[(size_t)j * PR]);
        *(v8bf*)(dst + idx) = d;
    }
}
// up_proj (n,r,h) -> upbT[n][h*128 + r], vector along r
__global__ void k_cvt_upT8(const float* __restrict__ up, __bf16* __restrict__ dst) {
    int i = blockIdx.x * 256 + threadIdx.x;
    if (i < NPROC * PR * HH / 8) {
        int idx = i * 8;
        int n = idx / (PR * HH), rem = idx % (PR * HH);
        int h = rem >> 7, r0 = rem & 127;
        const float* s = up + (size_t)n * PR * HH + (size_t)r0 * HH + h;
        v8bf d;
#pragma unroll
        for (int j = 0; j < 8; ++j) d[j] = f2bf(s[(size_t)j * HH]);
        *(v8bf*)(dst + idx) = d;
    }
}
// pattern inverse L2 norms
__global__ __launch_bounds__(256) void k_patnorm(const float* __restrict__ p, float* __restrict__ inv) {
    __shared__ float red[256];
    int n = blockIdx.x, tid = threadIdx.x;
    float s = 0.f;
    for (int j = tid; j < HH; j += 256) { float v = p[n * HH + j]; s += v * v; }
    red[tid] = s; __syncthreads();
    for (int o = 128; o > 0; o >>= 1) { if (tid < o) red[tid] += red[tid + o]; __syncthreads(); }
    if (tid == 0) inv[n] = 1.0f / fmaxf(sqrtf(red[0]), 1e-12f);
}

// ---------------- input_acts = sigmoid(l2norm(x) . l2norm(patterns)) ----------------
__global__ __launch_bounds__(256) void k_acts(const float* __restrict__ x, const float* __restrict__ patterns,
                                              const float* __restrict__ patInv, float* __restrict__ out_acts) {
    __shared__ float xr[HH];
    __shared__ float red[256];
    int row = blockIdx.x, tid = threadIdx.x;
    float ss = 0.f;
    for (int j = tid; j < HH; j += 256) { float v = x[row * HH + j]; xr[j] = v; ss += v * v; }
    red[tid] = ss; __syncthreads();
    for (int o = 128; o > 0; o >>= 1) { if (tid < o) red[tid] += red[tid + o]; __syncthreads(); }
    float invx = 1.0f / fmaxf(sqrtf(red[0]), 1e-12f);
    __syncthreads();
    int n = tid & 63, ch = tid >> 6;          // 4 chunks of 192
    const float* pr = patterns + n * HH + ch * 192;
    const float* xx = xr + ch * 192;
    float p = 0.f;
    for (int k = 0; k < 192; ++k) p += xx[k] * pr[k];
    red[tid] = p; __syncthreads();
    if (tid < 64) {
        float s = (red[tid] + red[tid + 64] + red[tid + 128] + red[tid + 192]) * invx * patInv[tid];
        out_acts[row * NIN + tid] = sigmoidf(s);
    }
}

// ---------------- LDS-free bf16 WMMA GEMM: C[M,N] = A[M,K] x BT[N,K]^T ----------------
// Both operands row-major along K -> every fragment is two global b128 loads; no LDS, no barriers.
// Each wave computes a 32x32 output (2x2 wmma tiles); block = 8 waves = 64(M) x 128(N).
// mode 0: store f32      mode 1: +bias, exact gelu, store f32 + bf16
// mode 2: scale by acts[row*64 + (col>>4)], store bf16
__global__ __launch_bounds__(256) void k_gemm(const __bf16* __restrict__ A,
                                              const __bf16* __restrict__ BT,
                                              float* __restrict__ Cf, __bf16* __restrict__ Cb,
                                              int M, int N, int K, int mode,
                                              const float* __restrict__ bias, const float* __restrict__ acts) {
    int n0 = blockIdx.x * 128, m0 = blockIdx.y * 64;
    int tid = threadIdx.x, w = tid >> 5, lane = tid & 31;
    int hi = lane >> 4, li = lane & 15;
    int wm = (w & 1) * 32, wn = (w >> 1) * 32;
    const __bf16* Ar0 = A  + (size_t)(m0 + wm + li) * K;
    const __bf16* Ar1 = A  + (size_t)(m0 + wm + 16 + li) * K;
    const __bf16* Br0 = BT + (size_t)(n0 + wn + li) * K;
    const __bf16* Br1 = BT + (size_t)(n0 + wn + 16 + li) * K;
    v8f a00 = {}, a01 = {}, a10 = {}, a11 = {};
    for (int k0 = 0; k0 < K; k0 += 32) {
        int o = k0 + 8 * hi;
        v16bf fa0 = ldfrag(Ar0 + o);
        v16bf fa1 = ldfrag(Ar1 + o);
        v16bf fb0 = ldfrag(Br0 + o);
        v16bf fb1 = ldfrag(Br1 + o);
        a00 = wmma_bf16(fa0, fb0, a00);
        a01 = wmma_bf16(fa0, fb1, a01);
        a10 = wmma_bf16(fa1, fb0, a10);
        a11 = wmma_bf16(fa1, fb1, a11);
    }
#pragma unroll
    for (int tm = 0; tm < 2; ++tm) {
#pragma unroll
        for (int tn = 0; tn < 2; ++tn) {
            v8f acc = tm ? (tn ? a11 : a10) : (tn ? a01 : a00);
#pragma unroll
            for (int r = 0; r < 8; ++r) {
                int row = m0 + wm + tm * 16 + r + 8 * hi;
                int col = n0 + wn + tn * 16 + li;
                float v = acc[r];
                if (mode == 1) { v += bias[col]; v = 0.5f * v * (1.0f + erff(v * 0.70710678118f)); }
                else if (mode == 2) { v *= acts[row * NIN + (col >> 4)]; }
                if (Cf) Cf[(size_t)row * N + col] = v;
                if (Cb) Cb[(size_t)row * N + col] = f2bf(v);
            }
        }
    }
}

// ---------------- x1 = LN(x + shared*actsum + adapterMix) ----------------
__global__ __launch_bounds__(256) void k_x1(const float* __restrict__ x, const float* __restrict__ interm,
                                            const float* __restrict__ sharedF, const float* __restrict__ acts,
                                            const float* __restrict__ g1, const float* __restrict__ b1,
                                            float* __restrict__ x1F, __bf16* __restrict__ x1b) {
    __shared__ float buf[HH];
    __shared__ float red[256];
    int row = blockIdx.x, tid = threadIdx.x;
    red[tid] = (tid < NIN) ? acts[row * NIN + tid] : 0.f;
    __syncthreads();
    for (int o = 128; o > 0; o >>= 1) { if (tid < o) red[tid] += red[tid + o]; __syncthreads(); }
    float actsum = red[0]; __syncthreads();
    float s1 = 0.f;
    for (int j = tid; j < HH; j += 256) {
        float v = x[row * HH + j] + interm[row * HH + j] + sharedF[row * HH + j] * actsum;
        buf[j] = v; s1 += v;
    }
    red[tid] = s1; __syncthreads();
    for (int o = 128; o > 0; o >>= 1) { if (tid < o) red[tid] += red[tid + o]; __syncthreads(); }
    float mean = red[0] / (float)HH; __syncthreads();
    float s2 = 0.f;
    for (int j = tid; j < HH; j += 256) { float d = buf[j] - mean; s2 += d * d; }
    red[tid] = s2; __syncthreads();
    for (int o = 128; o > 0; o >>= 1) { if (tid < o) red[tid] += red[tid + o]; __syncthreads(); }
    float rs = rsqrtf(red[0] / (float)HH + 1e-5f);
    for (int j = tid; j < HH; j += 256) {
        float y = (buf[j] - mean) * rs * g1[j] + b1[j];
        x1F[(size_t)row * HH + j] = y;
        x1b[(size_t)row * HH + j] = f2bf(y);
    }
}

// ---------------- qkv = acts @ in_proj^T + b ----------------
__global__ void k_qkv(const float* __restrict__ acts, const float* __restrict__ W,
                      const float* __restrict__ b, float* __restrict__ qkv) {
    int row = blockIdx.x, m = threadIdx.x;   // block 192
    float s = b[m];
    const float* a = acts + row * NIN;
    const float* wr = W + m * NIN;
    for (int n = 0; n < NIN; ++n) s += a[n] * wr[n];
    qkv[row * 192 + m] = s;
}

// ---------------- causal softmax attention (64-dim) ----------------
__global__ __launch_bounds__(256) void k_attn(const float* __restrict__ qkv,
                                              float* __restrict__ attn_out, float* __restrict__ ctx) {
    __shared__ float qv[NIN];
    __shared__ float wts[SS];
    __shared__ float red[256];
    int row = blockIdx.x, b = row >> 9, q = row & 511, tid = threadIdx.x;
    if (tid < NIN) qv[tid] = qkv[row * 192 + tid];
    __syncthreads();
    const float scale = 0.125f; // 1/sqrt(64)
    float lmax = NEG_INF;
    for (int j = tid; j < SS; j += 256) {
        float s = NEG_INF;
        if (j <= q) {
            const float* kv = qkv + ((size_t)((b << 9) + j)) * 192 + 64;
            float d = 0.f;
            for (int t = 0; t < NIN; ++t) d += qv[t] * kv[t];
            s = d * scale;
        }
        wts[j] = s;
        lmax = fmaxf(lmax, s);
    }
    red[tid] = lmax; __syncthreads();
    for (int o = 128; o > 0; o >>= 1) { if (tid < o) red[tid] = fmaxf(red[tid], red[tid + o]); __syncthreads(); }
    float mx = red[0]; __syncthreads();
    float lsum = 0.f;
    for (int j = tid; j < SS; j += 256) { float e = expf(wts[j] - mx); wts[j] = e; lsum += e; }
    red[tid] = lsum; __syncthreads();
    for (int o = 128; o > 0; o >>= 1) { if (tid < o) red[tid] += red[tid + o]; __syncthreads(); }
    float inv = 1.0f / red[0]; __syncthreads();
    for (int j = tid; j < SS; j += 256) {
        float wv = wts[j] * inv; wts[j] = wv;
        attn_out[((size_t)b * SS + q) * SS + j] = wv;
    }
    __syncthreads();
    if (tid < NIN) {
        float a = 0.f;
        for (int j = 0; j <= q; ++j) a += wts[j] * qkv[((size_t)((b << 9) + j)) * 192 + 128 + tid];
        ctx[row * NIN + tid] = a;
    }
}

// ---------------- relational + enriched ----------------
__global__ void k_rel(const float* __restrict__ ctx, const float* __restrict__ W, const float* __restrict__ b,
                      const float* __restrict__ acts, float* __restrict__ rel, float* __restrict__ enr) {
    int row = blockIdx.x, m = threadIdx.x;  // block 64
    float s = b[m];
    const float* c = ctx + row * NIN;
    const float* wr = W + m * NIN;
    for (int n = 0; n < NIN; ++n) s += c[n] * wr[n];
    rel[row * NIN + m] = s;
    enr[row * NIN + m] = s + acts[row * NIN + m];
}

// ---------------- conv (5 x 64 window -> 128 ch) + sigmoid ----------------
__global__ __launch_bounds__(128) void k_conv(const float* __restrict__ enr, const float* __restrict__ cw,
                                              const float* __restrict__ cb, float* __restrict__ pacts) {
    __shared__ float win[5 * NIN];
    int row = blockIdx.x, b = row >> 9, s = row & 511, tid = threadIdx.x;
    for (int e = tid; e < 5 * NIN; e += 128) {
        int dh = e >> 6, w = e & 63, sj = s + dh - 2;
        win[e] = (sj >= 0 && sj < SS) ? enr[((size_t)((b << 9) + sj)) * NIN + w] : 0.f;
    }
    __syncthreads();
    int o = tid;
    float acc = cb[o];
    const float* wr = cw + (size_t)o * 5 * NIN;
    for (int e = 0; e < 5 * NIN; ++e) acc += win[e] * wr[e];
    pacts[(size_t)row * NPROC + o] = sigmoidf(acc);
}

// ---------------- fused heavy path: sum_n pacts_n * ((x1 @ dp_n) @ up_n) ----------------
// block: 16 rows x 16 experts; grid (64 row-tiles, 8 expert-groups)
// Phase-1 fragments come straight from global (x1b row-major, dpbT[n][r][k]); only the
// cross-wave exchange (Pd) and the output accumulator live in LDS -> 2 barriers per expert.
__global__ __launch_bounds__(256) void k_heavy(const __bf16* __restrict__ x1b,
                                               const __bf16* __restrict__ dpbT,  // [n][r][k]
                                               const __bf16* __restrict__ upbT,  // [n][h][r]
                                               const float* __restrict__ pacts,
                                               float* __restrict__ partials) {
    __shared__ float accOut[16 * HH];              // 48 KB
    __shared__ __align__(16) __bf16 Pd[16 * 128];  // [m][r], 4 KB
    int s0 = blockIdx.x * 16, g = blockIdx.y;
    int tid = threadIdx.x, w = tid >> 5, lane = tid & 31;
    int hi = lane >> 4, li = lane & 15;
    for (int e = tid; e < 16 * HH; e += 256) accOut[e] = 0.f;
    __syncthreads();
    const __bf16* Ar = x1b + (size_t)(s0 + li) * HH;          // same A rows for all waves
    for (int nn = 0; nn < 16; ++nn) {
        int n = g * 16 + nn;
        // ---- phase 1: pdown(16x128) = x1(16x768) @ dp_n(768x128); wave w owns r-cols 16w..16w+15
        const __bf16* Br = dpbT + (size_t)n * PR * HH + (size_t)(w * 16 + li) * HH;
        v8f acc = {};
        for (int k0 = 0; k0 < HH; k0 += 32) {
            int o = k0 + 8 * hi;
            v16bf af = ldfrag(Ar + o);
            v16bf bf = ldfrag(Br + o);
            acc = wmma_bf16(af, bf, acc);
        }
        // scale by process_acts[s,n], repack bf16 into LDS
#pragma unroll
        for (int r = 0; r < 8; ++r) {
            int srow = s0 + r + 8 * hi;
            float v = acc[r] * pacts[(size_t)srow * NPROC + n];
            Pd[(r + 8 * hi) * 128 + w * 16 + li] = f2bf(v);
        }
        __syncthreads();
        // ---- phase 2: accOut(16x768) += pdown(16x128) @ up_n(128x768)
        const __bf16* Un = upbT + (size_t)n * PR * HH;   // [h][r]
        for (int ht = 0; ht < 6; ++ht) {
            int h0 = (w + 8 * ht) * 16;
            v8f acc2 = {};
#pragma unroll
            for (int kk = 0; kk < PR; kk += 32) {
                v16bf af2 = ldfrag(Pd + li * 128 + kk + 8 * hi);                // two ds_load_b128
                v16bf bf2 = ldfrag(Un + (size_t)(h0 + li) * 128 + kk + 8 * hi); // two global b128
                acc2 = wmma_bf16(af2, bf2, acc2);
            }
#pragma unroll
            for (int r = 0; r < 8; ++r)
                accOut[(r + 8 * hi) * HH + h0 + li] += acc2[r];  // unique (lane,wave) ownership
        }
        __syncthreads();
    }
    float* P = partials + (size_t)g * (BS * HH);
    for (int e = tid; e < 16 * HH; e += 256) {
        int r = e / HH, c = e % HH;
        P[(size_t)(s0 + r) * HH + c] = accOut[e];
    }
}

// ---------------- x2 = LN(x1 + sum_g partials) ----------------
__global__ __launch_bounds__(256) void k_final(const float* __restrict__ x1F, const float* __restrict__ partials,
                                               const float* __restrict__ g2, const float* __restrict__ b2,
                                               float* __restrict__ x2) {
    __shared__ float buf[HH];
    __shared__ float red[256];
    int row = blockIdx.x, tid = threadIdx.x;
    float s1 = 0.f;
    for (int j = tid; j < HH; j += 256) {
        float v = x1F[(size_t)row * HH + j];
        for (int gg = 0; gg < 8; ++gg) v += partials[(size_t)gg * BS * HH + (size_t)row * HH + j];
        buf[j] = v; s1 += v;
    }
    red[tid] = s1; __syncthreads();
    for (int o = 128; o > 0; o >>= 1) { if (tid < o) red[tid] += red[tid + o]; __syncthreads(); }
    float mean = red[0] / (float)HH; __syncthreads();
    float s2 = 0.f;
    for (int j = tid; j < HH; j += 256) { float d = buf[j] - mean; s2 += d * d; }
    red[tid] = s2; __syncthreads();
    for (int o = 128; o > 0; o >>= 1) { if (tid < o) red[tid] += red[tid + o]; __syncthreads(); }
    float rs = rsqrtf(red[0] / (float)HH + 1e-5f);
    for (int j = tid; j < HH; j += 256)
        x2[(size_t)row * HH + j] = (buf[j] - mean) * rs * g2[j] + b2[j];
}

// ---------------- host launch ----------------
extern "C" void kernel_launch(void* const* d_in, const int* in_sizes, int n_in,
                              void* d_out, int out_size, void* d_ws, size_t ws_size,
                              hipStream_t stream) {
    const float* x          = (const float*)d_in[0];
    const float* patterns   = (const float*)d_in[1];
    const float* w_shared   = (const float*)d_in[2];
    const float* b_shared   = (const float*)d_in[3];
    const float* adapt_down = (const float*)d_in[4];
    const float* adapt_up   = (const float*)d_in[5];
    const float* g1         = (const float*)d_in[6];
    const float* b1         = (const float*)d_in[7];
    const float* in_proj_w  = (const float*)d_in[8];
    const float* in_proj_b  = (const float*)d_in[9];
    const float* out_proj_w = (const float*)d_in[10];
    const float* out_proj_b = (const float*)d_in[11];
    const float* conv_w     = (const float*)d_in[12];
    const float* conv_b     = (const float*)d_in[13];
    const float* down_proj  = (const float*)d_in[14];
    const float* up_proj    = (const float*)d_in[15];
    const float* g2_        = (const float*)d_in[16];
    const float* b2_        = (const float*)d_in[17];

    // d_out layout (return order)
    float* out   = (float*)d_out;
    float* o_x2  = out;                       // 786432
    float* o_act = out + 786432;              // 65536
    float* o_rel = out + 786432 + 65536;      // 65536
    float* o_enr = out + 786432 + 131072;     // 65536
    float* o_pac = out + 786432 + 196608;     // 131072
    float* o_att = out + 786432 + 327680;     // 524288

    // workspace carve-up
    char* ws = (char*)d_ws;
    size_t off = 0;
    auto take = [&](size_t bytes) { char* p = ws + off; off = (off + bytes + 255) & ~(size_t)255; return p; };
    float*  patInv  = (float*)take(NIN * 4);
    __bf16* xb      = (__bf16*)take((size_t)BS * HH * 2);
    __bf16* wsB     = (__bf16*)take((size_t)HH * HH * 2);      // w_shared [o][k] (already B^T layout)
    __bf16* adBT    = (__bf16*)take((size_t)1024 * HH * 2);    // [(n*16+r)][k]
    __bf16* auBT    = (__bf16*)take((size_t)HH * 1024 * 2);    // [h][(n*16+r)]
    __bf16* dpbT    = (__bf16*)take((size_t)NPROC * PR * HH * 2); // [n][r][k]
    __bf16* upbT    = (__bf16*)take((size_t)NPROC * PR * HH * 2); // [n][h][r]
    float*  sharedF = (float*)take((size_t)BS * HH * 4);
    __bf16* sharedB = (__bf16*)take((size_t)BS * HH * 2);
    __bf16* downW   = (__bf16*)take((size_t)BS * 1024 * 2);
    float*  interm  = (float*)take((size_t)BS * HH * 4);
    float*  x1F     = (float*)take((size_t)BS * HH * 4);
    __bf16* x1b     = (__bf16*)take((size_t)BS * HH * 2);
    float*  qkv     = (float*)take((size_t)BS * 192 * 4);
    float*  ctx     = (float*)take((size_t)BS * NIN * 4);
    float*  parts   = (float*)take((size_t)8 * BS * HH * 4);
    (void)ws_size; (void)in_sizes; (void)n_in; (void)out_size;

    auto blocks8 = [](long n) { return (int)((n / 8 + 255) / 256); };

    // 1) operand prep (vectorized bf16 conversions; all B operands end up [n][k] row-major)
    k_cvt8<<<blocks8((long)BS * HH), 256, 0, stream>>>(x, xb, BS * HH / 8);
    k_cvt8<<<blocks8((long)HH * HH), 256, 0, stream>>>(w_shared, wsB, HH * HH / 8);   // (o,h) is already [n][k]
    k_cvt_adT8<<<blocks8((long)1024 * HH), 256, 0, stream>>>(adapt_down, adBT);
    k_cvtT8<<<blocks8((long)HH * 1024), 256, 0, stream>>>(adapt_up, auBT, HH, 1024);  // [h][(n,r)]
    k_cvt_dpT8<<<blocks8((long)NPROC * PR * HH), 256, 0, stream>>>(down_proj, dpbT);
    k_cvt_upT8<<<blocks8((long)NPROC * PR * HH), 256, 0, stream>>>(up_proj, upbT);
    k_patnorm<<<NIN, 256, 0, stream>>>(patterns, patInv);

    // 2) pattern activations
    k_acts<<<BS, 256, 0, stream>>>(x, patterns, patInv, o_act);

    // 3) shared = gelu(x @ W + b)   (M=1024,N=768,K=768)
    k_gemm<<<dim3(HH / 128, BS / 64), 256, 0, stream>>>(xb, wsB, sharedF, sharedB,
                                                        BS, HH, HH, 1, b_shared, nullptr);
    // 4) downW = (shared @ adapt_down) * acts   (M=1024,N=1024,K=768) -> bf16
    k_gemm<<<dim3(1024 / 128, BS / 64), 256, 0, stream>>>(sharedB, adBT, nullptr, downW,
                                                          BS, 1024, HH, 2, nullptr, o_act);
    // 5) interm = downW @ adapt_up   (M=1024,N=768,K=1024)
    k_gemm<<<dim3(HH / 128, BS / 64), 256, 0, stream>>>(downW, auBT, interm, nullptr,
                                                        BS, HH, 1024, 0, nullptr, nullptr);
    // 6) x1 = LN(x + shared*actsum + interm)
    k_x1<<<BS, 256, 0, stream>>>(x, interm, sharedF, o_act, g1, b1, x1F, x1b);

    // 7) attention side-path
    k_qkv<<<BS, 192, 0, stream>>>(o_act, in_proj_w, in_proj_b, qkv);
    k_attn<<<BS, 256, 0, stream>>>(qkv, o_att, ctx);
    k_rel<<<BS, NIN, 0, stream>>>(ctx, out_proj_w, out_proj_b, o_act, o_rel, o_enr);
    k_conv<<<BS, NPROC, 0, stream>>>(o_enr, conv_w, conv_b, o_pac);

    // 8) heavy expert path (WMMA fused)
    k_heavy<<<dim3(BS / 16, 8), 256, 0, stream>>>(x1b, dpbT, upbT, o_pac, parts);

    // 9) x2 = LN(x1 + sum partials)
    k_final<<<BS, 256, 0, stream>>>(x1F, parts, g2_, b2_, o_x2);
}
